// MentionScore_18700287607060
// MI455X (gfx1250) — compile-verified, matrix operand order checked by reading, not run on hardware
//
#include <hip/hip_runtime.h>

// ---------------------------------------------------------------------------
// Problem constants (from reference): B=8, T=4096, E=1024, S=1024, W=16, H=150
// ---------------------------------------------------------------------------
#define BB   8
#define TT   4096
#define EE   1024
#define SS   1024
#define WW   16
#define HH   150
#define HPAD 160          // H padded to 10 WMMA n-tiles of 16
#define NT   10           // number of 16-wide N tiles
#define FRAG_US 512       // ushorts per 32x16 fragment
#define STEP_US (NT * FRAG_US)   // 5120 ushorts staged per K-step

typedef __attribute__((ext_vector_type(16))) __bf16          v16bf;
typedef __attribute__((ext_vector_type(2)))  __bf16          v2bf;
typedef __attribute__((ext_vector_type(8)))  float           v8f;
typedef __attribute__((ext_vector_type(16))) unsigned short  v16us;

#if defined(__has_builtin)
#if __has_builtin(__builtin_amdgcn_cvt_pk_bf16_f32)
#define HAVE_PK_BF16 1
#endif
#endif

// fp32 -> bf16 with round-to-nearest-even (pack kernel / reference path)
static __device__ __forceinline__ unsigned short f2bf(float f) {
    unsigned u = __float_as_uint(f);
    u = (u + 0x7FFFu + ((u >> 16) & 1u)) >> 16;
    return (unsigned short)u;
}

// Convert 16 fp32 (four float4 groups) into a v16bf A-fragment.
// Prefer the packed-convert builtin; otherwise use __bf16 casts (fptrunc),
// which the gfx1250 backend can select to native bf16 converts.
static __device__ __forceinline__ v16bf pack16(const float4 g0, const float4 g1,
                                               const float4 g2, const float4 g3) {
#ifdef HAVE_PK_BF16
    union { v16bf v; v2bf p[8]; } u;
    u.p[0] = __builtin_amdgcn_cvt_pk_bf16_f32(g0.x, g0.y);
    u.p[1] = __builtin_amdgcn_cvt_pk_bf16_f32(g0.z, g0.w);
    u.p[2] = __builtin_amdgcn_cvt_pk_bf16_f32(g1.x, g1.y);
    u.p[3] = __builtin_amdgcn_cvt_pk_bf16_f32(g1.z, g1.w);
    u.p[4] = __builtin_amdgcn_cvt_pk_bf16_f32(g2.x, g2.y);
    u.p[5] = __builtin_amdgcn_cvt_pk_bf16_f32(g2.z, g2.w);
    u.p[6] = __builtin_amdgcn_cvt_pk_bf16_f32(g3.x, g3.y);
    u.p[7] = __builtin_amdgcn_cvt_pk_bf16_f32(g3.z, g3.w);
    return u.v;
#else
    v16bf a;
    a[0]  = (__bf16)g0.x; a[1]  = (__bf16)g0.y; a[2]  = (__bf16)g0.z; a[3]  = (__bf16)g0.w;
    a[4]  = (__bf16)g1.x; a[5]  = (__bf16)g1.y; a[6]  = (__bf16)g1.z; a[7]  = (__bf16)g1.w;
    a[8]  = (__bf16)g2.x; a[9]  = (__bf16)g2.y; a[10] = (__bf16)g2.z; a[11] = (__bf16)g2.w;
    a[12] = (__bf16)g3.x; a[13] = (__bf16)g3.y; a[14] = (__bf16)g3.z; a[15] = (__bf16)g3.w;
    return a;
#endif
}

// ---------------------------------------------------------------------------
// Pack fp32 weight matrix (Ksrc x 150, row-major) into bf16 WMMA B-fragments.
// Fragment f = kt*NT + nt holds a 32x16 (KxN) tile. Per CDNA5 16-bit operand
// layout: lane L owns column nt*16 + (L&15); its 16 elements cover
// K-locals {0..7,16..23} (L<16) or {8..15,24..31} (L>=16).
// dst index = f*512 + L*16 + e  (a lane's fragment is 32 contiguous bytes)
// ---------------------------------------------------------------------------
__global__ void pack_w1_frags(const float* __restrict__ src,
                              unsigned short* __restrict__ dst,
                              int kTiles, int Ksrc) {
    int t = blockIdx.x * blockDim.x + threadIdx.x;
    int total = kTiles * NT * FRAG_US;
    if (t >= total) return;
    int f   = t >> 9;
    int rem = t & 511;
    int L   = rem >> 4;
    int e   = rem & 15;
    int kt  = f / NT;
    int nt  = f - kt * NT;
    int n    = nt * 16 + (L & 15);
    int kloc = (e < 8 ? e : e + 8) + ((L >= 16) ? 8 : 0);
    int k    = kt * 32 + kloc;
    float v = (k < Ksrc && n < HH) ? src[(size_t)k * HH + n] : 0.0f;
    dst[t] = f2bf(v);
}

// ---------------------------------------------------------------------------
// GEMM: C(M x 160) = relu( A(M x lda, fp32) * W1(bf16 frags) + bias ), pad
// columns 150..159 with 0. One wave = 16 rows x 160 cols; block = 8 waves.
// Weight fragments for each K-step (10 KB) are staged once per block into
// LDS (double-buffered); all 10 fragments are pulled into registers before
// the WMMA burst so the 10 v_wmma issue back-to-back without per-op waits.
// ---------------------------------------------------------------------------
__global__ void __launch_bounds__(256)
gemm_bf16_relu(const float* __restrict__ A, int lda,
               const unsigned short* __restrict__ Bfrags,
               const float* __restrict__ bias,
               float* __restrict__ C) {
    __shared__ __align__(16) unsigned short bsh[2][STEP_US];

    const int tid  = threadIdx.x;
    const int wave = tid >> 5;
    const int lane = tid & 31;
    const int rowBase = blockIdx.x * 128 + wave * 16;
    const int kbase   = (lane >= 16) ? 8 : 0;     // K interleave offset
    const float* arow = A + (size_t)(rowBase + (lane & 15)) * lda;
    const int kTiles = lda >> 5;

    v8f acc[NT];
#pragma unroll
    for (int nt = 0; nt < NT; ++nt) acc[nt] = (v8f)(0.0f);

    // Stage K-step 0 weights into LDS buffer 0 (cooperative, uint4 granules).
    {
        const uint4* src = (const uint4*)Bfrags;
        uint4* dst = (uint4*)bsh[0];
        for (int i = tid; i < STEP_US / 8; i += 256) dst[i] = src[i];
    }
    __syncthreads();

    for (int kt = 0; kt < kTiles; ++kt) {
        const int cur = kt & 1;
        // Prefetch next K-step's weights into the other LDS buffer.
        if (kt + 1 < kTiles) {
            const uint4* src = (const uint4*)(Bfrags + (size_t)(kt + 1) * STEP_US);
            uint4* dst = (uint4*)bsh[cur ^ 1];
            for (int i = tid; i < STEP_US / 8; i += 256) dst[i] = src[i];
        }

        // A fragment: rows striped on lanes, K interleaved {0..7,16..23}/{8..15,24..31}
        const int k0 = kt * 32;
        const float4 g0 = *(const float4*)(arow + k0 + kbase);
        const float4 g1 = *(const float4*)(arow + k0 + kbase + 4);
        const float4 g2 = *(const float4*)(arow + k0 + kbase + 16);
        const float4 g3 = *(const float4*)(arow + k0 + kbase + 20);
        const v16bf a = pack16(g0, g1, g2, g3);

        // Pull all 10 B fragments into registers, then burst the WMMAs.
        const unsigned short* bk = &bsh[cur][lane * 16];
        v16bf bfr[NT];
#pragma unroll
        for (int nt = 0; nt < NT; ++nt)
            bfr[nt] = *(const v16bf*)(bk + nt * FRAG_US);
#pragma unroll
        for (int nt = 0; nt < NT; ++nt)
            acc[nt] = __builtin_amdgcn_wmma_f32_16x16x32_bf16(
                false, a, false, bfr[nt], (short)0, acc[nt], false, false);

        // All reads of bsh[cur] done before it is re-staged two steps later;
        // staging of bsh[cur^1] complete before the next iteration reads it.
        __syncthreads();
    }

    // Epilogue: C/D layout -> lane L holds col (L&15); VGPR r holds row r (+8 if L>=16)
    const int col   = lane & 15;
    const int rhalf = (lane >= 16) ? 8 : 0;
#pragma unroll
    for (int nt = 0; nt < NT; ++nt) {
        const int n = nt * 16 + col;
        const float bb = (n < HH) ? bias[n] : 0.0f;
#pragma unroll
        for (int r = 0; r < 8; ++r) {
            float v = acc[nt][r] + bb;
            v = fmaxf(v, 0.0f);
            if (n >= HH) v = 0.0f;                 // keep pad columns zero
            C[(size_t)(rowBase + r + rhalf) * HPAD + n] = v;
        }
    }
}

// ---------------------------------------------------------------------------
// Tail: h2 = relu(h1 * W2 + b2); score = h2 . w3 + b3. One block per row.
// h1 rows are 160 wide with zero padding, so only k<150 contributes anyway.
// ---------------------------------------------------------------------------
__global__ void __launch_bounds__(160)
mlp_tail(const float* __restrict__ H1, const float* __restrict__ w2,
         const float* __restrict__ b2, const float* __restrict__ w3,
         const float* __restrict__ b3, float* __restrict__ out) {
    __shared__ float h[HPAD];
    __shared__ float red[HPAD];
    const int j   = threadIdx.x;
    const int row = blockIdx.x;
    h[j] = H1[(size_t)row * HPAD + j];
    __syncthreads();
    float c = 0.0f;
    if (j < HH) {
        float acc = b2[j];
        for (int k = 0; k < HH; ++k)
            acc = fmaf(h[k], w2[k * HH + j], acc);
        acc = fmaxf(acc, 0.0f);
        c = acc * w3[j];
    }
    red[j] = c;
    __syncthreads();
    if (j == 0) {
        float s = b3[0];
        for (int k = 0; k < HH; ++k) s += red[k];
        out[row] = s;
    }
}

// ---------------------------------------------------------------------------
// Span gather: out[b,s] = concat(x[b,start], x[b,end], sum_w mask*att*x[b,start+w])
// One block (256 threads) per span; each thread owns 4 contiguous E columns.
// mask(w) == (w < width), so we just loop to width (uniform trip count).
// ---------------------------------------------------------------------------
__global__ void __launch_bounds__(256)
gather_weighted(const float* __restrict__ x, const int* __restrict__ starts,
                const int* __restrict__ ends, const float* __restrict__ attns,
                float* __restrict__ out) {
    const int bs = blockIdx.x;              // b*S + s
    const int b  = bs >> 10;                // S = 1024
    const int start = starts[bs];
    const int end   = ends[bs];
    const int width = end - start + 1;      // 1..16 (reference guarantees)
    const int e4 = threadIdx.x * 4;
    const float* xb = x + (size_t)b * TT * EE;

    const float4 se = *(const float4*)(xb + (size_t)start * EE + e4);
    const float4 ee = *(const float4*)(xb + (size_t)end   * EE + e4);
    float4 acc = make_float4(0.f, 0.f, 0.f, 0.f);
    for (int w = 0; w < width; ++w) {
        const int idx = start + w;
        const float a = attns[(size_t)b * TT + idx];
        const float4 t = *(const float4*)(xb + (size_t)idx * EE + e4);
        acc.x = fmaf(t.x, a, acc.x);
        acc.y = fmaf(t.y, a, acc.y);
        acc.z = fmaf(t.z, a, acc.z);
        acc.w = fmaf(t.w, a, acc.w);
    }
    float* o = out + (size_t)bs * (3 * EE) + e4;
    *(float4*)(o)            = se;
    *(float4*)(o + EE)       = ee;
    *(float4*)(o + 2 * EE)   = acc;
}

// ---------------------------------------------------------------------------
// Launcher
// ---------------------------------------------------------------------------
extern "C" void kernel_launch(void* const* d_in, const int* in_sizes, int n_in,
                              void* d_out, int out_size, void* d_ws, size_t ws_size,
                              hipStream_t stream) {
    const float* x      = (const float*)d_in[0];
    const int*   starts = (const int*)  d_in[1];
    const int*   ends   = (const int*)  d_in[2];
    const float* aw1 = (const float*)d_in[3];
    const float* ab1 = (const float*)d_in[4];
    const float* aw2 = (const float*)d_in[5];
    const float* ab2 = (const float*)d_in[6];
    const float* aw3 = (const float*)d_in[7];
    const float* ab3 = (const float*)d_in[8];
    const float* sw1 = (const float*)d_in[9];
    const float* sb1 = (const float*)d_in[10];
    const float* sw2 = (const float*)d_in[11];
    const float* sb2 = (const float*)d_in[12];
    const float* sw3 = (const float*)d_in[13];
    const float* sb3 = (const float*)d_in[14];
    float* out = (float*)d_out;

    // Workspace layout
    char* ws = (char*)d_ws;
    size_t off = 0;
    const int aKT = EE / 32;          // 32 K-tiles for attn layer 1
    const int sKT = (3 * EE) / 32;    // 96 K-tiles for span layer 1
    unsigned short* aw1f = (unsigned short*)(ws + off); off += (size_t)aKT * STEP_US * 2;
    unsigned short* sw1f = (unsigned short*)(ws + off); off += (size_t)sKT * STEP_US * 2;
    float* h1a   = (float*)(ws + off); off += (size_t)BB * TT * HPAD * 4;   // 32768 x 160
    float* h1s   = (float*)(ws + off); off += (size_t)BB * SS * HPAD * 4;   // 8192 x 160
    float* attns = (float*)(ws + off); off += (size_t)BB * TT * 4;          // 32768

    // 1) Pack layer-1 weights into bf16 WMMA fragments
    pack_w1_frags<<<(aKT * STEP_US + 255) / 256, 256, 0, stream>>>(aw1, aw1f, aKT, EE);
    pack_w1_frags<<<(sKT * STEP_US + 255) / 256, 256, 0, stream>>>(sw1, sw1f, sKT, 3 * EE);

    // 2) attn layer 1: (32768 x 1024) x (1024 x 150) -> h1a (relu)
    gemm_bf16_relu<<<(BB * TT) / 128, 256, 0, stream>>>(x, EE, aw1f, ab1, h1a);

    // 3) attn layers 2+3 -> per-token scores
    mlp_tail<<<BB * TT, HPAD, 0, stream>>>(h1a, aw2, ab2, aw3, ab3, attns);

    // 4) span gather + weighted sum -> span_embeds (first part of d_out)
    gather_weighted<<<BB * SS, 256, 0, stream>>>(x, starts, ends, attns, out);

    // 5) span layer 1: (8192 x 3072) x (3072 x 150) -> h1s (relu)
    gemm_bf16_relu<<<(BB * SS) / 128, 256, 0, stream>>>(out, 3 * EE, sw1f, sb1, h1s);

    // 6) span layers 2+3 -> mention scores (after span_embeds in d_out)
    mlp_tail<<<BB * SS, HPAD, 0, stream>>>(h1s, sw2, sb2, sw3, sb3,
                                           out + (size_t)BB * SS * 3 * EE);
}
